// UniversalToolIntegration_13288628814307
// MI455X (gfx1250) — matrix operand order repaired and checked by reading, not compiled, not run on hardware
//
#include <hip/hip_runtime.h>

// ---------------------------------------------------------------------------
// Problem constants (from reference)
// ---------------------------------------------------------------------------
#define N_TOOLS   50
#define DIM       512
#define KP        (DIM / 2)          // 256 bf16-pairs per row
#define N_TOKENS  8192
#define PARAM_DIM 256

// LDS pair-plane row stride: 260 % 64 == 4 -> the 16 lanes reading the same
// pair-column from different rows hit distinct banks; 260*4 % 16 == 0 keeps
// every row 16B aligned for ds_load_b128.
#define AP_STRIDE 260

// d_out layout (tuple flattened in return order, all f32)
#define OUT_IDX    ((size_t)0)
#define OUT_PROBS  ((size_t)N_TOKENS)
#define OUT_ADAPT  (OUT_PROBS + (size_t)N_TOKENS * N_TOOLS)
#define OUT_PARAMS (OUT_ADAPT + (size_t)N_TOKENS * DIM)

typedef __attribute__((ext_vector_type(16))) __bf16        v16bf;
typedef __attribute__((ext_vector_type(8)))  float         v8f;
typedef __attribute__((ext_vector_type(8)))  unsigned int  v8u;

// ---------------------------------------------------------------------------
// Workspace layout
//   ints   : tool_idx_i[8192] counts[64] offsets[65] cursor[64] perm[8192]
//   planes : rw (64 rows, zero-padded) | pw (256) | ew (50*512) ; hi+lo each
// ---------------------------------------------------------------------------
#define WS_INTS      16640                       // 66,560 bytes, 16B aligned
#define RW_ROWS_PAD  64
#define RW_PLANE     (RW_ROWS_PAD * KP)          // 16384 u32
#define PW_PLANE     (PARAM_DIM * KP)            // 65536 u32
#define EW_PLANE     ((size_t)N_TOOLS * DIM * KP)// 6,553,600 u32
#define WS_NEED_BYTES (WS_INTS * 4 + (2*RW_PLANE + 2*PW_PLANE + 2*EW_PLANE) * 4)

// ---------------------------------------------------------------------------
// bf16x2 split: f = hi(truncated bf16) + lo(RNE bf16 of residual)
// packed pair: [15:0] = elem0, [31:16] = elem1
// ---------------------------------------------------------------------------
__device__ __forceinline__ unsigned bf16_rne_bits(float f) {
  unsigned u = __float_as_uint(f);
  return (u + 0x7FFFu + ((u >> 16) & 1u)) >> 16;
}

__device__ __forceinline__ void split_pack(float f0, float f1,
                                           unsigned &hi, unsigned &lo) {
  unsigned u0 = __float_as_uint(f0) & 0xFFFF0000u;
  unsigned u1 = __float_as_uint(f1) & 0xFFFF0000u;
  hi = (u0 >> 16) | u1;
  lo = bf16_rne_bits(f0 - __uint_as_float(u0)) |
       (bf16_rne_bits(f1 - __uint_as_float(u1)) << 16);
}

__device__ __forceinline__ v16bf frag_from(uint4 a, uint4 b) {
  v8u u;
  u[0] = a.x; u[1] = a.y; u[2] = a.z; u[3] = a.w;
  u[4] = b.x; u[5] = b.y; u[6] = b.z; u[7] = b.w;
  return __builtin_bit_cast(v16bf, u);
}

// ---------------------------------------------------------------------------
// Fragment builders (contiguous k = 16*(L>>4) + s order, wave32)
// A 16x32: lane L row m=L&15; vgpr v holds pair-col k0/2 + 8*(L>>4)
//          + v  -> two b128 loads per plane
// ---------------------------------------------------------------------------
__device__ __forceinline__ void build_a_pre(const unsigned* Ahi,
                                            const unsigned* Alo,
                                            int lane, int k0,
                                            v16bf &ah, v16bf &al) {
  const int m  = lane & 15;
  const int c0 = (k0 >> 1) + ((lane >> 4) << 3);
  const uint4* ph = reinterpret_cast<const uint4*>(Ahi + m * AP_STRIDE + c0);
  const uint4* pl = reinterpret_cast<const uint4*>(Alo + m * AP_STRIDE + c0);
  ah = frag_from(ph[0], ph[1]);   // cols c0..c0+7
  al = frag_from(pl[0], pl[1]);
}

// B 32x16 (B[k][n] = W[n][k]): lane L col n=L&15; vgpr v holds pair-col
//          k0/2 + 8*(L>>4) + v  -> 8 consecutive u32 = two b128 per plane
__device__ __forceinline__ void build_b_pre(const unsigned* __restrict__ Whi,
                                            const unsigned* __restrict__ Wlo,
                                            int nbase, int lane, int k0,
                                            v16bf &bh, v16bf &bl) {
  const int n = nbase + (lane & 15);
  const size_t c = (size_t)n * KP + (k0 >> 1) + ((lane >> 4) << 3);
  const uint4* ph = reinterpret_cast<const uint4*>(Whi + c);
  const uint4* pl = reinterpret_cast<const uint4*>(Wlo + c);
  bh = frag_from(ph[0], ph[1]);
  bl = frag_from(pl[0], pl[1]);
}

// Fallback: split W on the fly, branchless (clamp row + mask multiply)
__device__ __forceinline__ void build_b_fly(const float* __restrict__ W,
                                            int nbase, int nmax,
                                            int lane, int k0,
                                            v16bf &bh, v16bf &bl) {
  int n = nbase + (lane & 15);
  const float msk = (n < nmax) ? 1.f : 0.f;
  n = (n < nmax) ? n : (nmax - 1);
  const int kh = (lane >> 4) << 4;
  v8u hi, lo;
#pragma unroll
  for (int v = 0; v < 8; ++v) {
    const int k = k0 + kh + (v << 1);
    const float2 f = *reinterpret_cast<const float2*>(W + (size_t)n * DIM + k);
    unsigned h, l;
    split_pack(f.x * msk, f.y * msk, h, l);
    hi[v] = h; lo[v] = l;
  }
  bh = __builtin_bit_cast(v16bf, hi);
  bl = __builtin_bit_cast(v16bf, lo);
}

// bf16x2 product: hi*hi + hi*lo + lo*hi (drop lo*lo, up to ~2^-14 relative)
__device__ __forceinline__ v8f wmma3(v16bf ah, v16bf al,
                                     v16bf bh, v16bf bl, v8f acc) {
  acc = __builtin_amdgcn_wmma_f32_16x16x32_bf16(false, ah, false, bh,
                                                (short)0, acc, false, false);
  acc = __builtin_amdgcn_wmma_f32_16x16x32_bf16(false, ah, false, bl,
                                                (short)0, acc, false, false);
  acc = __builtin_amdgcn_wmma_f32_16x16x32_bf16(false, al, false, bh,
                                                (short)0, acc, false, false);
  return acc;
}

// ---------------------------------------------------------------------------
// Kernel P: pre-split a weight matrix [rows][512] f32 into hi/lo pair planes
// rows >= rows_valid are written as zeros (router N-padding).
// ---------------------------------------------------------------------------
__global__ void split_weights_kernel(const float* __restrict__ W,
                                     unsigned* __restrict__ hi,
                                     unsigned* __restrict__ lo,
                                     int rows_alloc, int rows_valid) {
  const int i = blockIdx.x * blockDim.x + threadIdx.x;  // pair index
  if (i >= rows_alloc * KP) return;
  const int r = i >> 8;
  unsigned h = 0, l = 0;
  if (r < rows_valid) {
    const float2 f = reinterpret_cast<const float2*>(W)[i];
    split_pack(f.x, f.y, h, l);
  }
  hi[i] = h; lo[i] = l;
}

// ---------------------------------------------------------------------------
// Kernel 0: zero the atomic counters
// ---------------------------------------------------------------------------
__global__ void zero_counters_kernel(int* __restrict__ counts,
                                     int* __restrict__ cursor) {
  if (threadIdx.x < 64) { counts[threadIdx.x] = 0; cursor[threadIdx.x] = 0; }
}

// ---------------------------------------------------------------------------
// Kernel 1: router GEMM (16 tokens x 64 padded tools, K=512) + softmax +
// argmax + expert histogram. 128 threads = 4 waves, one 16-col tile each.
// ---------------------------------------------------------------------------
template <bool PRE>
__global__ __launch_bounds__(128)
void router_kernel(const float* __restrict__ x,
                   const float* __restrict__ rw,
                   const unsigned* __restrict__ rw_hi,
                   const unsigned* __restrict__ rw_lo,
                   const float* __restrict__ rb,
                   float* __restrict__ out,
                   int* __restrict__ tool_idx_i,
                   int* __restrict__ counts) {
  __shared__ unsigned Ahi[16 * AP_STRIDE];
  __shared__ unsigned Alo[16 * AP_STRIDE];
  __shared__ float    Lg[16 * 68];

  const int tokbase = blockIdx.x * 16;

  // stage + pre-split x tile (16 x 256 pairs)
  for (int i = threadIdx.x; i < 16 * KP; i += 128) {
    const int row = i >> 8, kp = i & 255;
    const float2 f =
        reinterpret_cast<const float2*>(x)[(size_t)(tokbase + row) * KP + kp];
    split_pack(f.x, f.y, Ahi[row * AP_STRIDE + kp], Alo[row * AP_STRIDE + kp]);
  }
  __syncthreads();

  const int lane  = threadIdx.x & 31;
  const int wave  = threadIdx.x >> 5;
  const int nbase = wave * 16;
  const int ncol  = nbase + (lane & 15);

  const float bias = (ncol < N_TOOLS) ? rb[ncol] : 0.f;
  v8f acc;
#pragma unroll
  for (int r = 0; r < 8; ++r) acc[r] = bias;

#pragma unroll
  for (int k0 = 0; k0 < DIM; k0 += 32) {
    v16bf ah, al, bh, bl;
    build_a_pre(Ahi, Alo, lane, k0, ah, al);
    if constexpr (PRE) build_b_pre(rw_hi, rw_lo, nbase, lane, k0, bh, bl);
    else               build_b_fly(rw, nbase, N_TOOLS, lane, k0, bh, bl);
    acc = wmma3(ah, al, bh, bl, acc);
  }

  // C layout: vgpr r -> (m = r + 8*(lane>>4), n = lane&15)
  const int mh = (lane >> 4) << 3;
#pragma unroll
  for (int r = 0; r < 8; ++r) Lg[(r + mh) * 68 + ncol] = acc[r];
  __syncthreads();

  if (threadIdx.x < 16) {
    const int t = tokbase + (int)threadIdx.x;
    const float* lrow = &Lg[threadIdx.x * 68];
    float mx = -3.402823466e38f; int bi = 0;
    for (int j = 0; j < N_TOOLS; ++j) {
      const float lv = lrow[j];
      if (lv > mx) { mx = lv; bi = j; }     // first-occurrence argmax
    }
    float s = 0.f;
    float* probs = out + OUT_PROBS + (size_t)t * N_TOOLS;
    for (int j = 0; j < N_TOOLS; ++j) {
      const float e = __expf(lrow[j] - mx);
      s += e; probs[j] = e;
    }
    const float inv = 1.f / s;
    for (int j = 0; j < N_TOOLS; ++j) probs[j] *= inv;
    out[OUT_IDX + t] = (float)bi;
    tool_idx_i[t] = bi;
    atomicAdd(&counts[bi], 1);
  }
}

// ---------------------------------------------------------------------------
// Kernel 2: exclusive prefix sum over expert counts (tiny)
// ---------------------------------------------------------------------------
__global__ void prefix_kernel(const int* __restrict__ counts,
                              int* __restrict__ offsets) {
  if (threadIdx.x == 0) {
    int acc = 0;
    for (int i = 0; i < 64; ++i) { offsets[i] = acc; acc += counts[i]; }
    offsets[64] = acc;
  }
}

// ---------------------------------------------------------------------------
// Kernel 3: scatter token ids into per-expert segments
// ---------------------------------------------------------------------------
__global__ void scatter_kernel(const int* __restrict__ tool_idx_i,
                               const int* __restrict__ offsets,
                               int* __restrict__ cursor,
                               int* __restrict__ perm) {
  const int t = blockIdx.x * blockDim.x + threadIdx.x;
  if (t < N_TOKENS) {
    const int e = tool_idx_i[t];
    const int p = atomicAdd(&cursor[e], 1);
    perm[offsets[e] + p] = t;
  }
}

// ---------------------------------------------------------------------------
// Kernel 4: gathered per-expert GEMM. grid=(50,64); 256 threads = 8 waves;
// tile = 16 gathered tokens x 512 cols (8 waves x 16 cols x 4 passes), K=512.
// ---------------------------------------------------------------------------
template <bool PRE>
__global__ __launch_bounds__(256)
void expert_kernel(const float* __restrict__ x,
                   const float* __restrict__ ew,
                   const unsigned* __restrict__ ew_hi,
                   const unsigned* __restrict__ ew_lo,
                   const float* __restrict__ eb,
                   const int* __restrict__ offsets,
                   const int* __restrict__ perm,
                   float* __restrict__ out) {
  __shared__ unsigned Ahi[16 * AP_STRIDE];
  __shared__ unsigned Alo[16 * AP_STRIDE];
  __shared__ int      toks[16];

  const int e   = blockIdx.x;
  const int beg = offsets[e];
  const int cnt = offsets[e + 1] - beg;
  const float*    Wf  = ew + (size_t)e * DIM * DIM;
  const unsigned* Whi = ew_hi + (size_t)e * DIM * KP;
  const unsigned* Wlo = ew_lo + (size_t)e * DIM * KP;

  const int lane = threadIdx.x & 31;
  const int wave = threadIdx.x >> 5;
  float* adapt = out + OUT_ADAPT;

  for (int tile = blockIdx.y; tile * 16 < cnt; tile += gridDim.y) {
    if (threadIdx.x < 16) {
      const int idx = tile * 16 + (int)threadIdx.x;
      toks[threadIdx.x] = (idx < cnt) ? perm[beg + idx] : -1;
    }
    __syncthreads();

    for (int i = threadIdx.x; i < 16 * KP; i += 256) {
      const int row = i >> 8, kp = i & 255;
      const int t = toks[row];
      float2 f = make_float2(0.f, 0.f);
      if (t >= 0) f = reinterpret_cast<const float2*>(x)[(size_t)t * KP + kp];
      split_pack(f.x, f.y, Ahi[row * AP_STRIDE + kp],
                 Alo[row * AP_STRIDE + kp]);
    }
    __syncthreads();

#pragma unroll
    for (int pass = 0; pass < 4; ++pass) {
      const int nbase = pass * 128 + wave * 16;
      const int ncol  = nbase + (lane & 15);
      const float bias = eb[(size_t)e * DIM + ncol];
      v8f acc;
#pragma unroll
      for (int r = 0; r < 8; ++r) acc[r] = bias;

#pragma unroll
      for (int k0 = 0; k0 < DIM; k0 += 32) {
        v16bf ah, al, bh, bl;
        build_a_pre(Ahi, Alo, lane, k0, ah, al);
        if constexpr (PRE) build_b_pre(Whi, Wlo, nbase, lane, k0, bh, bl);
        else               build_b_fly(Wf, nbase, DIM, lane, k0, bh, bl);
        acc = wmma3(ah, al, bh, bl, acc);
      }

      const int mh = (lane >> 4) << 3;
#pragma unroll
      for (int r = 0; r < 8; ++r) {
        const int t = toks[r + mh];
        if (t >= 0) adapt[(size_t)t * DIM + ncol] = acc[r];
      }
    }
    __syncthreads();
  }
}

// ---------------------------------------------------------------------------
// Kernel 5: param head GEMM [8192x512]*[512x256]^T + bias
// ---------------------------------------------------------------------------
template <bool PRE>
__global__ __launch_bounds__(256)
void param_kernel(const float* __restrict__ x,
                  const float* __restrict__ pw,
                  const unsigned* __restrict__ pw_hi,
                  const unsigned* __restrict__ pw_lo,
                  const float* __restrict__ pb,
                  float* __restrict__ out) {
  __shared__ unsigned Ahi[16 * AP_STRIDE];
  __shared__ unsigned Alo[16 * AP_STRIDE];

  const int tokbase = blockIdx.x * 16;

  for (int i = threadIdx.x; i < 16 * KP; i += 256) {
    const int row = i >> 8, kp = i & 255;
    const float2 f =
        reinterpret_cast<const float2*>(x)[(size_t)(tokbase + row) * KP + kp];
    split_pack(f.x, f.y, Ahi[row * AP_STRIDE + kp], Alo[row * AP_STRIDE + kp]);
  }
  __syncthreads();

  const int lane = threadIdx.x & 31;
  const int wave = threadIdx.x >> 5;
  float* params = out + OUT_PARAMS;

#pragma unroll
  for (int pass = 0; pass < 2; ++pass) {
    const int nbase = pass * 128 + wave * 16;
    const int ncol  = nbase + (lane & 15);
    const float bias = pb[ncol];
    v8f acc;
#pragma unroll
    for (int r = 0; r < 8; ++r) acc[r] = bias;

#pragma unroll
    for (int k0 = 0; k0 < DIM; k0 += 32) {
      v16bf ah, al, bh, bl;
      build_a_pre(Ahi, Alo, lane, k0, ah, al);
      if constexpr (PRE) build_b_pre(pw_hi, pw_lo, nbase, lane, k0, bh, bl);
      else               build_b_fly(pw, nbase, PARAM_DIM, lane, k0, bh, bl);
      acc = wmma3(ah, al, bh, bl, acc);
    }

    const int mh = (lane >> 4) << 3;
#pragma unroll
    for (int r = 0; r < 8; ++r) {
      const int t = tokbase + r + mh;
      params[(size_t)t * PARAM_DIM + ncol] = acc[r];
    }
  }
}

// ---------------------------------------------------------------------------
// Launcher
// ---------------------------------------------------------------------------
extern "C" void kernel_launch(void* const* d_in, const int* in_sizes, int n_in,
                              void* d_out, int out_size, void* d_ws,
                              size_t ws_size, hipStream_t stream) {
  const float* x        = (const float*)d_in[0];
  const float* router_w = (const float*)d_in[1];
  const float* router_b = (const float*)d_in[2];
  const float* expert_w = (const float*)d_in[3];
  const float* expert_b = (const float*)d_in[4];
  const float* param_w  = (const float*)d_in[5];
  const float* param_b  = (const float*)d_in[6];
  float* out = (float*)d_out;

  // workspace: int region then split-weight planes
  int* wsi        = (int*)d_ws;
  int* tool_idx_i = wsi;             // 8192
  int* counts     = wsi + 8192;      // 64
  int* offsets    = wsi + 8256;      // 65
  int* cursor     = wsi + 8328;      // 64
  int* perm       = wsi + 8392;      // 8192

  unsigned* planes = (unsigned*)d_ws + WS_INTS;
  unsigned* rw_hi = planes;
  unsigned* rw_lo = rw_hi + RW_PLANE;
  unsigned* pw_hi = rw_lo + RW_PLANE;
  unsigned* pw_lo = pw_hi + PW_PLANE;
  unsigned* ew_hi = pw_lo + PW_PLANE;
  unsigned* ew_lo = ew_hi + EW_PLANE;

  const bool pre = (ws_size >= (size_t)WS_NEED_BYTES);

  zero_counters_kernel<<<1, 128, 0, stream>>>(counts, cursor);

  if (pre) {
    split_weights_kernel<<<(RW_ROWS_PAD * KP) / 256, 256, 0, stream>>>(
        router_w, rw_hi, rw_lo, RW_ROWS_PAD, N_TOOLS);
    split_weights_kernel<<<(PARAM_DIM * KP) / 256, 256, 0, stream>>>(
        param_w, pw_hi, pw_lo, PARAM_DIM, PARAM_DIM);
    split_weights_kernel<<<(N_TOOLS * DIM * KP) / 256, 256, 0, stream>>>(
        expert_w, ew_hi, ew_lo, N_TOOLS * DIM, N_TOOLS * DIM);

    router_kernel<true><<<N_TOKENS / 16, 128, 0, stream>>>(
        x, router_w, rw_hi, rw_lo, router_b, out, tool_idx_i, counts);
    prefix_kernel<<<1, 1, 0, stream>>>(counts, offsets);
    scatter_kernel<<<N_TOKENS / 256, 256, 0, stream>>>(
        tool_idx_i, offsets, cursor, perm);
    expert_kernel<true><<<dim3(N_TOOLS, 64), 256, 0, stream>>>(
        x, expert_w, ew_hi, ew_lo, expert_b, offsets, perm, out);
    param_kernel<true><<<N_TOKENS / 16, 256, 0, stream>>>(
        x, param_w, pw_hi, pw_lo, param_b, out);
  } else {
    router_kernel<false><<<N_TOKENS / 16, 128, 0, stream>>>(
        x, router_w, rw_hi, rw_lo, router_b, out, tool_idx_i, counts);
    prefix_kernel<<<1, 1, 0, stream>>>(counts, offsets);
    scatter_kernel<<<N_TOKENS / 256, 256, 0, stream>>>(
        tool_idx_i, offsets, cursor, perm);
    expert_kernel<false><<<dim3(N_TOOLS, 64), 256, 0, stream>>>(
        x, expert_w, ew_hi, ew_lo, expert_b, offsets, perm, out);
    param_kernel<false><<<N_TOKENS / 16, 256, 0, stream>>>(
        x, param_w, pw_hi, pw_lo, param_b, out);
  }
}
